// PointNet2Seg_8813272891498
// MI455X (gfx1250) — matrix operand order repaired and checked by reading, not compiled, not run on hardware
//
#include <hip/hip_runtime.h>
#include <hip/hip_bf16.h>
#include <math.h>

typedef _Float16 f16;
typedef __attribute__((ext_vector_type(16))) _Float16 v16h;
typedef __attribute__((ext_vector_type(8)))  _Float16 v8h;
typedef __attribute__((ext_vector_type(8)))  float    v8f;

// ---------------------------------------------------------------------------
// Weight prep: W f32 [O,C] -> Wp f16 [Opad, Cpad], zero padded.
// ---------------------------------------------------------------------------
__global__ void prep_w_kernel(const float* __restrict__ W, f16* __restrict__ Wp,
                              int O, int C, int Opad, int Cpad) {
  int i = blockIdx.x * blockDim.x + threadIdx.x;
  int total = Opad * Cpad;
  if (i >= total) return;
  int o = i / Cpad, c = i - o * Cpad;
  Wp[i] = (o < O && c < C) ? (f16)W[(size_t)o * C + c] : (f16)0.f;
}

__global__ void prep_b_kernel(const float* __restrict__ b, float* __restrict__ bp,
                              int O, int Opad) {
  int o = blockIdx.x * blockDim.x + threadIdx.x;
  if (o < Opad) bp[o] = (o < O) ? b[o] : 0.f;
}

// ---------------------------------------------------------------------------
// WMMA GEMM, guard-free hot loop.
//   X : f16 [N, Cpad]   (positions x channels, pad channels are zero)
//   Wp: f16 [Opad, Cpad]
//   Y : f16 [N, Opad]   Y[n,o] = sum_c X[n,c]*Wp[o,c] + bp[o]
// A fragment = X tile (16 pos x 32 ch), B fragment = Wp tile (32 ch x 16 o).
// One wave per 16x16 tile; 4 waves/block cover 4 o-tiles; dual accumulators.
// ---------------------------------------------------------------------------
__global__ void gemm_wmma_kernel(const f16* __restrict__ X,
                                 const f16* __restrict__ Wp,
                                 const float* __restrict__ bp,
                                 f16* __restrict__ Y,
                                 int Cpad, int N, int Opad) {
  const int n0 = blockIdx.x * 16;
  const int o0 = blockIdx.y * 64 + threadIdx.y * 16;
  if (o0 >= Opad) return;
  const int lane = threadIdx.x;          // 0..31
  const int m    = lane & 15;
  const bool hi  = lane >= 16;
  const int kA   = hi ? 8 : 0;           // A 16-bit layout: lo K0-7/16-23, hi K8-15/24-31
  const int kB   = hi ? 8 : 0;           // B 16-bit layout: lo K0-7/16-23, hi K8-15/24-31

  const f16* xrow = X  + (size_t)(n0 + m) * Cpad;   // A: position row
  const f16* wrow = Wp + (size_t)(o0 + m) * Cpad;   // B: weight row

  v8f acc0 = {}, acc1 = {};
  int k0 = 0;
  for (; k0 + 64 <= Cpad; k0 += 64) {
    __builtin_prefetch(xrow + k0 + 128, 0, 1);      // global_prefetch_b8
    {
      v8h a0 = *(const v8h*)(xrow + k0 + kA);
      v8h a1 = *(const v8h*)(xrow + k0 + kA + 16);
      v8h b0 = *(const v8h*)(wrow + k0 + kB);
      v8h b1 = *(const v8h*)(wrow + k0 + kB + 16);
      v16h a, bm;
#pragma unroll
      for (int e = 0; e < 8; ++e) { a[e] = a0[e]; a[8 + e] = a1[e]; }
#pragma unroll
      for (int e = 0; e < 8; ++e) { bm[e] = b0[e]; bm[8 + e] = b1[e]; }
      acc0 = __builtin_amdgcn_wmma_f32_16x16x32_f16(false, a, false, bm,
                                                    (short)0, acc0, false, false);
    }
    {
      int k1 = k0 + 32;
      v8h a0 = *(const v8h*)(xrow + k1 + kA);
      v8h a1 = *(const v8h*)(xrow + k1 + kA + 16);
      v8h b0 = *(const v8h*)(wrow + k1 + kB);
      v8h b1 = *(const v8h*)(wrow + k1 + kB + 16);
      v16h a, bm;
#pragma unroll
      for (int e = 0; e < 8; ++e) { a[e] = a0[e]; a[8 + e] = a1[e]; }
#pragma unroll
      for (int e = 0; e < 8; ++e) { bm[e] = b0[e]; bm[8 + e] = b1[e]; }
      acc1 = __builtin_amdgcn_wmma_f32_16x16x32_f16(false, a, false, bm,
                                                    (short)0, acc1, false, false);
    }
  }
  if (k0 < Cpad) {                                   // Cpad % 64 == 32 tail
    v8h a0 = *(const v8h*)(xrow + k0 + kA);
    v8h a1 = *(const v8h*)(xrow + k0 + kA + 16);
    v8h b0 = *(const v8h*)(wrow + k0 + kB);
    v8h b1 = *(const v8h*)(wrow + k0 + kB + 16);
    v16h a, bm;
#pragma unroll
    for (int e = 0; e < 8; ++e) { a[e] = a0[e]; a[8 + e] = a1[e]; }
#pragma unroll
    for (int e = 0; e < 8; ++e) { bm[e] = b0[e]; bm[8 + e] = b1[e]; }
    acc0 = __builtin_amdgcn_wmma_f32_16x16x32_f16(false, a, false, bm,
                                                  (short)0, acc0, false, false);
  }

  const float bias = bp[o0 + m];
#pragma unroll
  for (int r = 0; r < 8; ++r) {
    int pos = n0 + (hi ? 8 + r : r);     // D: VGPR r -> M=r (lo lanes) / 8+r (hi)
    Y[(size_t)pos * Opad + o0 + m] = (f16)(acc0[r] + acc1[r] + bias);
  }
}

// ---------------------------------------------------------------------------
// BN stats, deterministic two-stage. Y: [N, Opad]; one thread per (o, chunk).
// partial[(o*64+chunk)*2 + {0,1}] = {sum, sumsq}
// ---------------------------------------------------------------------------
__global__ void bn_stats_kernel(const f16* __restrict__ Y,
                                float* __restrict__ partial,
                                int N, int O, int Opad) {
  int o = blockIdx.y * blockDim.x + threadIdx.x;
  if (o >= O) return;
  int chunk = blockIdx.x;                           // 0..63
  float s = 0.f, s2 = 0.f;
  for (int n = chunk; n < N; n += 64) {
    float v = (float)Y[(size_t)n * Opad + o];
    s += v; s2 += v * v;
  }
  partial[((size_t)o * 64 + chunk) * 2 + 0] = s;
  partial[((size_t)o * 64 + chunk) * 2 + 1] = s2;
}

__global__ void bn_finalize_kernel(const float* __restrict__ partial,
                                   float* __restrict__ stats, int O) {
  int o = blockIdx.x * blockDim.x + threadIdx.x;
  if (o >= O) return;
  float s = 0.f, s2 = 0.f;
  for (int j = 0; j < 64; ++j) {
    s  += partial[((size_t)o * 64 + j) * 2 + 0];
    s2 += partial[((size_t)o * 64 + j) * 2 + 1];
  }
  stats[2 * o + 0] = s;
  stats[2 * o + 1] = s2;
}

// normalize + scale + ReLU; writes zeros into pad channels (o >= O).
__global__ void bn_relu_kernel(const f16* __restrict__ Y,
                               const float* __restrict__ stats,
                               const float* __restrict__ g,
                               const float* __restrict__ bt,
                               f16* __restrict__ Xout, int N, int O, int Opad) {
  size_t i = (size_t)blockIdx.x * blockDim.x + threadIdx.x;
  size_t total = (size_t)N * Opad;
  if (i >= total) return;
  int o = (int)(i % (size_t)Opad);
  float v = 0.f;
  if (o < O) {
    float invN = 1.f / (float)N;
    float mean = stats[2 * o] * invN;
    float var  = stats[2 * o + 1] * invN - mean * mean;
    v = ((float)Y[i] - mean) * rsqrtf(var + 1e-5f);
    v = fmaxf(v * g[o] + bt[o], 0.f);
  }
  Xout[i] = (f16)v;
}

// ---------------------------------------------------------------------------
// Centers: idx = floor(m*(P-1)/(M-1))  (jnp.linspace + floor)
// ---------------------------------------------------------------------------
__global__ void gather_centers_kernel(const float* __restrict__ src,
                                      float* __restrict__ dst,
                                      int B, int P, int M) {
  int i = blockIdx.x * blockDim.x + threadIdx.x;
  if (i >= B * M) return;
  int b = i / M, m = i - b * M;
  int idx = (int)floorf((float)m * (float)(P - 1) / (float)(M - 1));
  if (idx > P - 1) idx = P - 1;
  const float* s = src + ((size_t)b * P + idx) * 3;
  float* d = dst + (size_t)i * 3;
  d[0] = s[0]; d[1] = s[1]; d[2] = s[2];
}

// ---------------------------------------------------------------------------
// KNN: one wave per query; distances in LDS; K rounds of lane-parallel argmin.
// blockDim = (32, 2); dynamic LDS = 2*Pr floats.
// ---------------------------------------------------------------------------
__global__ void knn_kernel(const float* __restrict__ query,
                           const float* __restrict__ ref,
                           int* __restrict__ outIdx, float* __restrict__ outD2,
                           int B, int Mq, int Pr, int K) {
  extern __shared__ float distLds[];
  const int lane = threadIdx.x;
  const int wave = threadIdx.y;
  float* d = distLds + (size_t)wave * Pr;
  int q = blockIdx.x * blockDim.y + wave;
  if (q >= B * Mq) return;
  int b = q / Mq;
  const float* qp = query + (size_t)q * 3;
  float qx = qp[0], qy = qp[1], qz = qp[2];
  const float* rp = ref + (size_t)b * Pr * 3;
  for (int p = lane; p < Pr; p += 32) {
    float dx = qx - rp[p * 3 + 0];
    float dy = qy - rp[p * 3 + 1];
    float dz = qz - rp[p * 3 + 2];
    d[p] = dx * dx + dy * dy + dz * dz;
  }
  for (int j = 0; j < K; ++j) {
    float best = 3.4e38f; int bestIdx = 0x7fffffff;
    for (int p = lane; p < Pr; p += 32) {
      float v = d[p];
      if (v < best || (v == best && p < bestIdx)) { best = v; bestIdx = p; }
    }
    for (int off = 16; off > 0; off >>= 1) {
      float ov = __shfl_down(best, off, 32);
      int   oi = __shfl_down(bestIdx, off, 32);
      if (ov < best || (ov == best && oi < bestIdx)) { best = ov; bestIdx = oi; }
    }
    bestIdx = __shfl(bestIdx, 0, 32);
    best    = __shfl(best, 0, 32);
    if (lane == 0) {
      outIdx[(size_t)q * K + j] = bestIdx;
      if (outD2) outD2[(size_t)q * K + j] = best;
      d[bestIdx] = 3.4e38f;
    }
  }
}

// ---------------------------------------------------------------------------
// SA input: Xcat[(b,m,kk), c] ; c<3 -> neigh-center ; 3..Ccat -> feats gather.
// feats: [B*Pr, Cf] position-major. Pad channels (c >= 3+Cf) zeroed.
// ---------------------------------------------------------------------------
__global__ void build_cat_kernel(const float* __restrict__ refXyz,
                                 const float* __restrict__ centers,
                                 const int* __restrict__ knnIdx,
                                 const f16* __restrict__ feats,
                                 f16* __restrict__ Xcat,
                                 int B, int M, int Pr, int K, int Cf, int Cpad) {
  size_t N = (size_t)B * M * K;
  size_t total = N * (size_t)Cpad;
  size_t i = (size_t)blockIdx.x * blockDim.x + threadIdx.x;
  if (i >= total) return;
  size_t n = i / (size_t)Cpad;
  int c = (int)(i - n * (size_t)Cpad);
  int Ccat = 3 + Cf;
  float v = 0.f;
  if (c < Ccat) {
    int kk = (int)(n % (size_t)K);
    size_t g = n / (size_t)K;
    int b = (int)(g / (size_t)M);
    int m = (int)(g - (size_t)b * M);
    int pidx = knnIdx[g * K + kk];
    if (c < 3)
      v = refXyz[((size_t)b * Pr + pidx) * 3 + c] -
          centers[((size_t)b * M + m) * 3 + c];
    else
      v = (float)feats[((size_t)b * Pr + pidx) * Cf + (c - 3)];
  }
  Xcat[i] = (f16)v;
}

// max over K contiguous positions: X [G*K, C] -> out [G, C]
__global__ void maxpool_kernel(const f16* __restrict__ X, f16* __restrict__ out,
                               int C, int G, int K) {
  size_t total = (size_t)G * C;
  size_t i = (size_t)blockIdx.x * blockDim.x + threadIdx.x;
  if (i >= total) return;
  size_t g = i / (size_t)C;
  int c = (int)(i - g * (size_t)C);
  const f16* p = X + g * (size_t)K * C + c;
  float mx = -3.4e38f;
  for (int kk = 0; kk < K; ++kk) mx = fmaxf(mx, (float)p[(size_t)kk * C]);
  out[i] = (f16)mx;
}

// 3-NN inverse-distance interp into channels [0,C) of out rows (stride Cstr).
// feats2: [B*Mref, C] position-major.
__global__ void interp3_kernel(const int* __restrict__ idx3,
                               const float* __restrict__ d2,
                               const f16* __restrict__ feats2,
                               f16* __restrict__ out,
                               int C, int Cstr, int B, int Mq, int Mref) {
  size_t N = (size_t)B * Mq;
  size_t total = N * (size_t)C;
  size_t i = (size_t)blockIdx.x * blockDim.x + threadIdx.x;
  if (i >= total) return;
  size_t n = i / (size_t)C;
  int c = (int)(i - n * (size_t)C);
  int b = (int)(n / (size_t)Mq);
  float w0 = 1.f / fmaxf(sqrtf(d2[n * 3 + 0]), 1e-8f);
  float w1 = 1.f / fmaxf(sqrtf(d2[n * 3 + 1]), 1e-8f);
  float w2 = 1.f / fmaxf(sqrtf(d2[n * 3 + 2]), 1e-8f);
  float s = w0 + w1 + w2;
  w0 /= s; w1 /= s; w2 /= s;
  const f16* fb = feats2 + (size_t)b * Mref * C;
  float v = w0 * (float)fb[(size_t)idx3[n * 3 + 0] * C + c] +
            w1 * (float)fb[(size_t)idx3[n * 3 + 1] * C + c] +
            w2 * (float)fb[(size_t)idx3[n * 3 + 2] * C + c];
  out[n * (size_t)Cstr + c] = (f16)v;
}

// append f1 [N,Cf] into out rows at channel offset coff (row stride Cstr)
__global__ void copy_f1_kernel(const f16* __restrict__ src, f16* __restrict__ dst,
                               int Cf, int Cstr, int coff, int N) {
  size_t total = (size_t)N * Cf;
  size_t i = (size_t)blockIdx.x * blockDim.x + threadIdx.x;
  if (i >= total) return;
  size_t n = i / (size_t)Cf;
  int cf = (int)(i - n * (size_t)Cf);
  dst[n * (size_t)Cstr + coff + cf] = src[i];
}

// logits: Y [N,16] f16 -> out [B,L,10] f32 (n = b*L+l)
__global__ void final_out_kernel(const f16* __restrict__ Y,
                                 float* __restrict__ out, int Opad, int O,
                                 int total) {
  int i = blockIdx.x * blockDim.x + threadIdx.x;
  if (i >= total) return;
  int o = i % O;
  int n = i / O;
  out[i] = (float)Y[(size_t)n * Opad + o];
}

// ---------------------------------------------------------------------------
extern "C" void kernel_launch(void* const* d_in, const int* in_sizes, int n_in,
                              void* d_out, int out_size, void* d_ws, size_t ws_size,
                              hipStream_t stream) {
  (void)in_sizes; (void)ws_size;
  if (n_in < 71) return;
  const int B = 16, P0 = 4096;
  const float* xyz = (const float*)d_in[0];
  auto P = [&](int i) { return (const float*)d_in[i]; };

  // ---- workspace carve ----
  char* wp = (char*)d_ws;
  auto carve = [&](size_t bytes) {
    char* r = wp;
    wp += (bytes + 255) & ~(size_t)255;
    return (void*)r;
  };
  const size_t MAXX = (size_t)131072 * 1024;        // largest [N,Cpad]
  f16*   bufA     = (f16*)carve(MAXX * sizeof(f16));
  f16*   bufB     = (f16*)carve(MAXX * sizeof(f16));
  float* centers1 = (float*)carve((size_t)B * 1024 * 3 * 4);
  float* centers2 = (float*)carve((size_t)B * 256 * 3 * 4);
  float* centers3 = (float*)carve((size_t)B * 64 * 3 * 4);
  int*   knnIdx   = (int*)carve((size_t)524288 * 4);
  float* knnD2    = (float*)carve((size_t)524288 * 4);
  f16*   l1  = (f16*)carve((size_t)B * 1024 * 128 * sizeof(f16));
  f16*   l2  = (f16*)carve((size_t)B * 256 * 256 * sizeof(f16));
  f16*   l3  = (f16*)carve((size_t)B * 64 * 1024 * sizeof(f16));
  f16*   l2n = (f16*)carve((size_t)B * 256 * 256 * sizeof(f16));
  f16*   l1n = (f16*)carve((size_t)B * 1024 * 128 * sizeof(f16));
  f16*   Wp  = (f16*)carve((size_t)1024 * 1280 * sizeof(f16));
  float* bp  = (float*)carve((size_t)1024 * 4);
  float* partials = (float*)carve((size_t)1024 * 64 * 2 * 4);
  float* stats    = (float*)carve((size_t)2048 * 4);

  auto knn = [&](const float* q, const float* r, int* oi, float* od2,
                 int Mq, int Pr, int K) {
    dim3 blk(32, 2);
    int total = B * Mq;
    knn_kernel<<<dim3((unsigned)((total + 1) / 2)), blk, (size_t)2 * Pr * 4,
                 stream>>>(q, r, oi, od2, B, Mq, Pr, K);
  };

  // GEMM (+ optional global BN + ReLU). pi -> {W, b, g, bt}; Cpad = Xin stride.
  auto layer = [&](const f16* Xin, int Cpad, int Creal, f16* Ytmp, f16* Xout,
                   int O, int N, int pi, bool bnrelu) {
    int Opad = (O + 15) & ~15;
    prep_w_kernel<<<(Opad * Cpad + 255) / 256, 256, 0, stream>>>(
        P(pi), Wp, O, Creal, Opad, Cpad);
    prep_b_kernel<<<(Opad + 63) / 64, 64, 0, stream>>>(P(pi + 1), bp, O, Opad);
    dim3 g1((unsigned)(N / 16), (unsigned)((Opad + 63) / 64));
    gemm_wmma_kernel<<<g1, dim3(32, 4), 0, stream>>>(Xin, Wp, bp, Ytmp,
                                                     Cpad, N, Opad);
    if (bnrelu) {
      bn_stats_kernel<<<dim3(64, (unsigned)((O + 255) / 256)), 256, 0, stream>>>(
          Ytmp, partials, N, O, Opad);
      bn_finalize_kernel<<<(O + 63) / 64, 64, 0, stream>>>(partials, stats, O);
      size_t total = (size_t)N * Opad;
      bn_relu_kernel<<<(unsigned)((total + 255) / 256), 256, 0, stream>>>(
          Ytmp, stats, P(pi + 2), P(pi + 3), Xout, N, O, Opad);
    }
  };

  auto cat_sa = [&](const float* refXyz, const float* ctr, const f16* feats,
                    int M, int Pr, int K, int Cf, int Cpad) {
    size_t tot = (size_t)B * M * K * Cpad;
    build_cat_kernel<<<(unsigned)((tot + 255) / 256), 256, 0, stream>>>(
        refXyz, ctr, knnIdx, feats, bufA, B, M, Pr, K, Cf, Cpad);
  };

  // ================= SA1 =================
  gather_centers_kernel<<<(B * 1024 + 255) / 256, 256, 0, stream>>>(
      xyz, centers1, B, P0, 1024);
  knn(centers1, xyz, knnIdx, nullptr, 1024, P0, 32);
  cat_sa(xyz, centers1, nullptr, 1024, P0, 32, 0, 32);
  const int N1 = B * 1024 * 32;                     // 524288
  layer(bufA, 32, 3, bufB, bufA, 64, N1, 1, true);
  layer(bufA, 64, 64, bufB, bufA, 64, N1, 5, true);
  layer(bufA, 64, 64, bufB, bufA, 128, N1, 9, true);
  maxpool_kernel<<<(unsigned)(((size_t)B * 1024 * 128 + 255) / 256), 256, 0,
                   stream>>>(bufA, l1, 128, B * 1024, 32);

  // ================= SA2 =================
  gather_centers_kernel<<<(B * 256 + 255) / 256, 256, 0, stream>>>(
      centers1, centers2, B, 1024, 256);
  knn(centers2, centers1, knnIdx, nullptr, 256, 1024, 64);
  cat_sa(centers1, centers2, l1, 256, 1024, 64, 128, 160);
  const int N2 = B * 256 * 64;                      // 262144
  layer(bufA, 160, 131, bufB, bufA, 128, N2, 13, true);
  layer(bufA, 128, 128, bufB, bufA, 128, N2, 17, true);
  layer(bufA, 128, 128, bufB, bufA, 256, N2, 21, true);
  maxpool_kernel<<<(unsigned)(((size_t)B * 256 * 256 + 255) / 256), 256, 0,
                   stream>>>(bufA, l2, 256, B * 256, 64);

  // ================= SA3 =================
  gather_centers_kernel<<<(B * 64 + 255) / 256, 256, 0, stream>>>(
      centers2, centers3, B, 256, 64);
  knn(centers3, centers2, knnIdx, nullptr, 64, 256, 128);
  cat_sa(centers2, centers3, l2, 64, 256, 128, 256, 288);
  const int N3 = B * 64 * 128;                      // 131072
  layer(bufA, 288, 259, bufB, bufA, 256, N3, 25, true);
  layer(bufA, 256, 256, bufB, bufA, 512, N3, 29, true);
  layer(bufA, 512, 512, bufB, bufA, 1024, N3, 33, true);
  maxpool_kernel<<<(unsigned)(((size_t)B * 64 * 1024 + 255) / 256), 256, 0,
                   stream>>>(bufA, l3, 1024, B * 64, 128);

  // ================= FP3 =================
  knn(centers2, centers3, knnIdx, knnD2, 256, 64, 3);
  {
    size_t ti = (size_t)B * 256 * 1024;
    interp3_kernel<<<(unsigned)((ti + 255) / 256), 256, 0, stream>>>(
        knnIdx, knnD2, l3, bufA, 1024, 1280, B, 256, 64);
    copy_f1_kernel<<<(unsigned)(((size_t)B * 256 * 256 + 255) / 256), 256, 0,
                     stream>>>(l2, bufA, 256, 1280, 1024, B * 256);
  }
  const int Nf3 = B * 256;                          // 4096
  layer(bufA, 1280, 1280, bufB, bufA, 256, Nf3, 37, true);
  layer(bufA, 256, 256, bufB, l2n, 256, Nf3, 41, true);

  // ================= FP2 =================
  knn(centers1, centers2, knnIdx, knnD2, 1024, 256, 3);
  {
    size_t ti = (size_t)B * 1024 * 256;
    interp3_kernel<<<(unsigned)((ti + 255) / 256), 256, 0, stream>>>(
        knnIdx, knnD2, l2n, bufA, 256, 384, B, 1024, 256);
    copy_f1_kernel<<<(unsigned)(((size_t)B * 1024 * 128 + 255) / 256), 256, 0,
                     stream>>>(l1, bufA, 128, 384, 256, B * 1024);
  }
  const int Nf2 = B * 1024;                         // 16384
  layer(bufA, 384, 384, bufB, bufA, 256, Nf2, 45, true);
  layer(bufA, 256, 256, bufB, l1n, 128, Nf2, 49, true);

  // ================= FP1 =================
  knn(xyz, centers1, knnIdx, knnD2, 4096, 1024, 3);
  {
    size_t ti = (size_t)B * 4096 * 128;
    interp3_kernel<<<(unsigned)((ti + 255) / 256), 256, 0, stream>>>(
        knnIdx, knnD2, l1n, bufA, 128, 128, B, 4096, 1024);
  }
  const int Nf1 = B * 4096;                         // 65536
  layer(bufA, 128, 128, bufB, bufA, 128, Nf1, 53, true);
  layer(bufA, 128, 128, bufB, bufA, 128, Nf1, 57, true);
  layer(bufA, 128, 128, bufB, bufA, 128, Nf1, 61, true);

  // ================= heads =================
  layer(bufA, 128, 128, bufB, bufA, 128, Nf1, 65, true);   // head1
  layer(bufA, 128, 128, bufB, nullptr, 10, Nf1, 69, false); // head2: linear only
  final_out_kernel<<<(out_size + 255) / 256, 256, 0, stream>>>(
      bufB, (float*)d_out, 16, 10, out_size);
}
